// MixtralNotPagedAttn_45114336477572
// MI455X (gfx1250) — matrix operand order repaired and checked by reading, not compiled, not run on hardware
//
#include <hip/hip_runtime.h>
#include <hip/hip_bf16.h>

typedef __attribute__((ext_vector_type(16))) _Float16 v16h;
typedef __attribute__((ext_vector_type(8)))  _Float16 v8h;
typedef __attribute__((ext_vector_type(8)))  float    v8f;

#define SEQ_LEN    2048
#define HIDDEN_DIM 4096
#define N_HEADS    32
#define N_KVH      8
#define HDIM       128
#define N_GROUPS   4
#define KV_STRIDE  (N_KVH * HDIM)   // 1024

__device__ __forceinline__ float hmax16(float v) {
#pragma unroll
  for (int m = 1; m < 16; m <<= 1) v = fmaxf(v, __shfl_xor(v, m, 16));
  return v;
}
__device__ __forceinline__ float hsum16(float v) {
#pragma unroll
  for (int m = 1; m < 16; m <<= 1) v += __shfl_xor(v, m, 16);
  return v;
}

// Block = 256 threads = 8 waves: 4 heads of one KV group x 2 q-subtiles (32 q rows).
// K/V tile (32 keys) double-buffered in LDS; global loads for tile kt+1 issued
// before the WMMA work of tile kt, converted/stored after it (latency hidden).
__launch_bounds__(256, 1)
__global__ void gqa_flash_fwd_kernel(const float* __restrict__ Qg,
                                     const float* __restrict__ Kg,
                                     const float* __restrict__ Vg,
                                     float* __restrict__ Og) {
  const int qt   = blockIdx.x;       // 0..63 : 32-row q super-tile
  const int kvh  = blockIdx.y;       // 0..7  : KV head
  const int tid  = (int)threadIdx.x; // 0..255
  const int wave = tid >> 5;         // 0..7
  const int lane = tid & 31;
  const int g    = lane >> 4;        // half-wave select
  const int ln   = lane & 15;
  const int hg   = wave & 3;         // head within group
  const int qsub = wave >> 2;        // 0..1 : q-subtile
  const int head = kvh * N_GROUPS + hg;
  const int q0   = qt * 32 + qsub * 16;

  __shared__ __align__(32) _Float16 qbuf[8][16 * HDIM];     // 32 KB (q * scale, f16)
  __shared__ __align__(32) _Float16 kbuf[2][32 * HDIM];     // 16 KB [buf][key][d]
  __shared__ __align__(32) _Float16 vtbuf[2][HDIM * 32];    // 16 KB [buf][d][key]
  __shared__ __align__(32) _Float16 pbuf[8][16 * 32];       // 8 KB

  const float scale = 0.08838834764831845f;  // 1/sqrt(128), folded into Q

  // ---- stage this wave's Q tile (fp32 -> f16, pre-scaled) ----
  {
    const float* src = Qg + (size_t)q0 * HIDDEN_DIM + head * HDIM;
#pragma unroll
    for (int i = 0; i < 16; ++i) {
      int idx = (lane + i * 32) * 4;           // 512 float4 / 32 lanes
      int row = idx >> 7, col = idx & 127;
      float4 f = *(const float4*)(src + (size_t)row * HIDDEN_DIM + col);
      _Float16* d = &qbuf[wave][row * HDIM + col];
      d[0] = (_Float16)(f.x * scale); d[1] = (_Float16)(f.y * scale);
      d[2] = (_Float16)(f.z * scale); d[3] = (_Float16)(f.w * scale);
    }
  }
  asm volatile("s_wait_dscnt 0x0" ::: "memory");

  // ---- preload Q A-fragments: 4 chunks of K=32 over d=128 ----
  // A layout (16-bit 16x32): lane row m=ln, runs at K = 8g..8g+7 and 16+8g..23+8g
  v16h qfrag[4];
  {
    const _Float16* qrow = &qbuf[wave][ln * HDIM];
#pragma unroll
    for (int c = 0; c < 4; ++c) {
      const int base = c * 32 + 8 * g;
      v8h lo = *(const v8h*)(qrow + base);
      v8h hi = *(const v8h*)(qrow + base + 16);
      v16h a;
#pragma unroll
      for (int e = 0; e < 8; ++e) { a[e] = lo[e]; a[e + 8] = hi[e]; }
      qfrag[c] = a;
    }
  }

  v8f acc[8];
#pragma unroll
  for (int t = 0; t < 8; ++t) { v8f z = {}; acc[t] = z; }
  float rmax[8], rsum[8];
#pragma unroll
  for (int r = 0; r < 8; ++r) { rmax[r] = -1.0e30f; rsum[r] = 0.0f; }

  const int ktiles = qt + 1;  // identical for all 8 waves (32-row supertile)

  const float* kbase = Kg + (size_t)kvh * HDIM;
  const float* vbase = Vg + (size_t)kvh * HDIM;
  const int vrow = tid & 31;          // key row for V staging
  const int vcb  = (tid >> 5) * 16;   // d block for V staging

  float4 kreg[4], vreg[4];

  auto load_tile = [&](int kt) {
    const int k0 = kt * 32;
    const float* ks = kbase + (size_t)k0 * KV_STRIDE;
#pragma unroll
    for (int i = 0; i < 4; ++i) {
      int idx = (tid + i * 256) * 4;          // 1024 float4 / 256 threads
      int row = idx >> 7, col = idx & 127;
      kreg[i] = *(const float4*)(ks + (size_t)row * KV_STRIDE + col);
    }
    const float* vs = vbase + (size_t)k0 * KV_STRIDE;
#pragma unroll
    for (int j = 0; j < 4; ++j) {
      vreg[j] = *(const float4*)(vs + (size_t)vrow * KV_STRIDE + vcb + j * 4);
    }
    if (kt + 1 < ktiles) {  // prefetch the tile after next (global_prefetch_b8)
      const float* nk = kbase + (size_t)(k0 + 32) * KV_STRIDE;
      const float* nv = vbase + (size_t)(k0 + 32) * KV_STRIDE;
      int half = (tid >> 5) & 1;
      __builtin_prefetch(nk + (size_t)vrow * KV_STRIDE + half * 64, 0, 1);
      __builtin_prefetch(nv + (size_t)vrow * KV_STRIDE + half * 64, 0, 1);
    }
  };
  auto store_tile = [&](int buf) {
#pragma unroll
    for (int i = 0; i < 4; ++i) {
      int idx = (tid + i * 256) * 4;
      int row = idx >> 7, col = idx & 127;
      _Float16* d = &kbuf[buf][row * HDIM + col];
      d[0] = (_Float16)kreg[i].x; d[1] = (_Float16)kreg[i].y;
      d[2] = (_Float16)kreg[i].z; d[3] = (_Float16)kreg[i].w;
    }
#pragma unroll
    for (int j = 0; j < 4; ++j) {
      int col = vcb + j * 4;
      vtbuf[buf][(col + 0) * 32 + vrow] = (_Float16)vreg[j].x;
      vtbuf[buf][(col + 1) * 32 + vrow] = (_Float16)vreg[j].y;
      vtbuf[buf][(col + 2) * 32 + vrow] = (_Float16)vreg[j].z;
      vtbuf[buf][(col + 3) * 32 + vrow] = (_Float16)vreg[j].w;
    }
  };

  load_tile(0);
  store_tile(0);

  for (int kt = 0; kt < ktiles; ++kt) {
    __syncthreads();                       // buf[kt&1] staged by all waves
    const int buf = kt & 1;
    const int k0  = kt * 32;
    const bool have_next = (kt + 1) < ktiles;
    if (have_next) load_tile(kt + 1);      // issue global loads, consume later

    const bool diag    = (kt == ktiles - 1);
    const bool skip_hi = diag && (qsub == 0);  // upper 16 keys fully masked

    // ---- S = Q K^T : two 16-key N-tiles, 4 K-chunks each ----
    // B layout (32x16): lane col n=ln, runs at K = 8g..8g+7 and 16+8g..23+8g
    v8f s0 = {}, s1 = {};
#pragma unroll
    for (int c = 0; c < 4; ++c) {
      const _Float16* kp0 = &kbuf[buf][(0 * 16 + ln) * HDIM + c * 32 + 8 * g];
      v8h kl0 = *(const v8h*)kp0, kh0 = *(const v8h*)(kp0 + 16);
      v16h b0 = __builtin_shufflevector(kl0, kh0, 0,1,2,3,4,5,6,7,8,9,10,11,12,13,14,15);
      s0 = __builtin_amdgcn_wmma_f32_16x16x32_f16(false, qfrag[c], false, b0,
                                                  (short)0, s0, false, false);
    }
    if (!skip_hi) {
#pragma unroll
      for (int c = 0; c < 4; ++c) {
        const _Float16* kp1 = &kbuf[buf][(1 * 16 + ln) * HDIM + c * 32 + 8 * g];
        v8h kl1 = *(const v8h*)kp1, kh1 = *(const v8h*)(kp1 + 16);
        v16h b1 = __builtin_shufflevector(kl1, kh1, 0,1,2,3,4,5,6,7,8,9,10,11,12,13,14,15);
        s1 = __builtin_amdgcn_wmma_f32_16x16x32_f16(false, qfrag[c], false, b1,
                                                    (short)0, s1, false, false);
      }
    }

    // ---- causal mask (scale already folded into Q) ----
    if (diag) {
#pragma unroll
      for (int r = 0; r < 8; ++r) {
        int qrow = q0 + r + 8 * g;
        if (k0 + ln > qrow)      s0[r] += -1.0e9f;
        if (k0 + 16 + ln > qrow) s1[r] += -1.0e9f;
      }
    }

    // ---- online softmax (row reductions across each half-wave) ----
    float pr0[8], pr1[8];
#pragma unroll
    for (int r = 0; r < 8; ++r) {
      float mcur  = hmax16(fmaxf(s0[r], s1[r]));
      float mnew  = fmaxf(rmax[r], mcur);
      float alpha = __expf(rmax[r] - mnew);
      float p0 = __expf(s0[r] - mnew);
      float p1 = __expf(s1[r] - mnew);
      float ps = hsum16(p0 + p1);
      rsum[r] = rsum[r] * alpha + ps;
      rmax[r] = mnew;
#pragma unroll
      for (int t = 0; t < 8; ++t) acc[t][r] *= alpha;
      pr0[r] = p0; pr1[r] = p1;
    }

    // ---- P -> LDS (C layout) -> reload as A fragment (intra-wave) ----
#pragma unroll
    for (int r = 0; r < 8; ++r) {
      int m = r + 8 * g;
      pbuf[wave][m * 32 + ln]      = (_Float16)pr0[r];
      pbuf[wave][m * 32 + 16 + ln] = (_Float16)pr1[r];
    }
    asm volatile("s_wait_dscnt 0x0" ::: "memory");
    v16h pf;
    {
      const _Float16* pr = &pbuf[wave][ln * 32 + 8 * g];
      v8h lo = *(const v8h*)pr;
      v8h hi = *(const v8h*)(pr + 16);
#pragma unroll
      for (int e = 0; e < 8; ++e) { pf[e] = lo[e]; pf[e + 8] = hi[e]; }
    }

    // ---- O += P V : 8 d-column tiles ----
#pragma unroll
    for (int t = 0; t < 8; ++t) {
      const _Float16* vp = &vtbuf[buf][(t * 16 + ln) * 32 + 8 * g];
      v8h vl = *(const v8h*)vp, vh = *(const v8h*)(vp + 16);
      v16h b = __builtin_shufflevector(vl, vh, 0,1,2,3,4,5,6,7,8,9,10,11,12,13,14,15);
      acc[t] = __builtin_amdgcn_wmma_f32_16x16x32_f16(false, pf, false, b,
                                                      (short)0, acc[t], false, false);
    }

    if (have_next) store_tile((kt + 1) & 1);  // waits for loads here, after compute
  }

  // ---- epilogue: normalize and store (out[s, head*128 + d]) ----
#pragma unroll
  for (int r = 0; r < 8; ++r) rsum[r] = 1.0f / rsum[r];
  float* obase = Og + (size_t)head * HDIM;
#pragma unroll
  for (int t = 0; t < 8; ++t) {
#pragma unroll
    for (int r = 0; r < 8; ++r) {
      int m = r + 8 * g;
      int d = t * 16 + ln;
      obase[(size_t)(q0 + m) * HIDDEN_DIM + d] = acc[t][r] * rsum[r];
    }
  }
}

// past_key / past_value: [S, kvh*128+d] -> [kvh][S][d]
__global__ void kvcache_copy_kernel(const float* __restrict__ Kg,
                                    const float* __restrict__ Vg,
                                    float* __restrict__ pk,
                                    float* __restrict__ pv) {
  int i = blockIdx.x * blockDim.x + (int)threadIdx.x;
  const int total = N_KVH * SEQ_LEN * HDIM;  // 2,097,152
  if (i >= total) return;
  int d = i & (HDIM - 1);
  int s = (i >> 7) & (SEQ_LEN - 1);
  int h = i >> 18;
  size_t src = (size_t)s * KV_STRIDE + h * HDIM + d;
  pk[i] = Kg[src];
  pv[i] = Vg[src];
}

extern "C" void kernel_launch(void* const* d_in, const int* in_sizes, int n_in,
                              void* d_out, int out_size, void* d_ws, size_t ws_size,
                              hipStream_t stream) {
  (void)in_sizes; (void)n_in; (void)out_size; (void)d_ws; (void)ws_size;
  const float* Q = (const float*)d_in[0];
  const float* K = (const float*)d_in[1];
  const float* V = (const float*)d_in[2];
  // d_in[3] (attention_mask) is the standard additive causal mask; computed in-kernel.

  float* attn = (float*)d_out;                                   // 2048*4096
  float* pk   = attn + (size_t)SEQ_LEN * HIDDEN_DIM;             // 8*2048*128
  float* pv   = pk + (size_t)N_KVH * SEQ_LEN * HDIM;

  dim3 grid(SEQ_LEN / 32, N_KVH);
  gqa_flash_fwd_kernel<<<grid, 256, 0, stream>>>(Q, K, V, attn);

  const int total = N_KVH * SEQ_LEN * HDIM;
  kvcache_copy_kernel<<<(total + 255) / 256, 256, 0, stream>>>(K, V, pk, pv);
}
